// Block_76519137345684
// MI455X (gfx1250) — compile-verified, matrix operand order checked by reading, not run on hardware
//
#include <hip/hip_runtime.h>

// ---------------------------------------------------------------------------
// MI455X (gfx1250) Spikformer block.
// bf16 WMMA for the 6 big GEMMs (4x4 register-blocked waves, BN stats fused
// via f32 atomics), f16 WMMA for binary-spike attention (bit-exact),
// elementwise BN+LIF scans.
// ---------------------------------------------------------------------------

typedef __attribute__((ext_vector_type(4)))  unsigned int u32x4;
typedef __attribute__((ext_vector_type(16))) __bf16      bf16x16;
typedef __attribute__((ext_vector_type(16))) _Float16    f16x16;
typedef __attribute__((ext_vector_type(8)))  float       f32x8;

union FragBF { bf16x16 v; u32x4 q[2]; };
union FragHF { f16x16  v; u32x4 q[2]; };

#define EPSV 1e-5f

// ---------------- fp32 -> bf16 convert ----------------
__global__ void k_f32_to_bf16(const float* __restrict__ s, __bf16* __restrict__ d, int n) {
  int i = blockIdx.x * blockDim.x + threadIdx.x;
  if (i < n) d[i] = (__bf16)s[i];
}

// ---------------- bf16 WMMA GEMM: Y[M,O] = A[M,K] @ W[O,K]^T + bias ---------
// Block: 128 threads = 4 waves (2 M-waves x 2 O-waves), wave tile 64x64
// (4x4 wmma 16x16x32) -> 16 WMMAs per 8 fragment loads per K-step.
// Block tile 128x128. Fused per-channel sum/sumsq atomics for BatchNorm.
__global__ __launch_bounds__(128)
void k_gemm_bf16(const __bf16* __restrict__ A, const __bf16* __restrict__ W,
                 const float* __restrict__ bias, float* __restrict__ Y,
                 float* __restrict__ stats, int M, int K, int O) {
  const int lane = threadIdx.x & 31;
  const int wave = threadIdx.x >> 5;  // 0..3
  const int lm   = lane & 15;
  const int half = lane >> 4;         // K-split half per ISA A/B layout
  const int wm   = wave >> 1;         // 0..1
  const int wo   = wave & 1;          // 0..1
  const int mBase = blockIdx.x * 128 + wm * 64;
  const int oBase = blockIdx.y * 128 + wo * 64;

  f32x8 acc[4][4] = {};

  const __bf16* ap[4];
  const __bf16* bp[4];
  #pragma unroll
  for (int i = 0; i < 4; i++) {
    ap[i] = A + (size_t)(mBase + i * 16 + lm) * K + half * 8;
    bp[i] = W + (size_t)(oBase + i * 16 + lm) * K + half * 8;
  }

  for (int k0 = 0; k0 < K; k0 += 32) {
    FragBF a[4], b[4];
    #pragma unroll
    for (int i = 0; i < 4; i++) {
      a[i].q[0] = *(const u32x4*)(ap[i] + k0);
      a[i].q[1] = *(const u32x4*)(ap[i] + k0 + 16);
      b[i].q[0] = *(const u32x4*)(bp[i] + k0);
      b[i].q[1] = *(const u32x4*)(bp[i] + k0 + 16);
    }
    if (k0 + 32 < K) {                       // keep next K-slab warm in cache
      __builtin_prefetch(ap[0] + k0 + 32, 0, 1);
      __builtin_prefetch(bp[0] + k0 + 32, 0, 1);
    }
    #pragma unroll
    for (int i = 0; i < 4; i++) {
      #pragma unroll
      for (int j = 0; j < 4; j++) {
        acc[i][j] = __builtin_amdgcn_wmma_f32_16x16x32_bf16(
            false, a[i].v, false, b[j].v, (short)0, acc[i][j], false, false);
      }
    }
  }

  #pragma unroll
  for (int i = 0; i < 4; i++) {
    #pragma unroll
    for (int j = 0; j < 4; j++) {
      const int o  = oBase + j * 16 + lm;
      const int r0 = mBase + i * 16 + half * 8;
      const float bv = bias[o];
      float s1 = 0.f, s2 = 0.f;
      #pragma unroll
      for (int r = 0; r < 8; r++) {
        float v = acc[i][j][r] + bv;
        Y[(size_t)(r0 + r) * O + o] = v;
        s1 += v; s2 += v * v;
      }
      atomicAdd(&stats[o],     s1);
      atomicAdd(&stats[O + o], s2);
    }
  }
}

// ---------------- BN stats finalize: scale/shift per channel ----------------
__global__ void k_finalize(const float* __restrict__ stats, const float* __restrict__ g,
                           const float* __restrict__ be, float* __restrict__ scl,
                           float* __restrict__ sft, int O, float invM) {
  int i = blockIdx.x * blockDim.x + threadIdx.x;
  if (i >= O) return;
  float mu  = stats[i] * invM;
  float var = stats[O + i] * invM - mu * mu;
  float sc  = rsqrtf(var + EPSV) * g[i];
  scl[i] = sc;
  sft[i] = be[i] - mu * sc;
}

// ---------------- BN (optional) + LIF scan over T=4 -------------------------
template <typename TO>
__global__ void k_bn_lif(const float* __restrict__ Y, const float* __restrict__ scl,
                         const float* __restrict__ sft, TO* __restrict__ out,
                         int O, float th, int total) {
  int i = blockIdx.x * blockDim.x + threadIdx.x;
  if (i >= total) return;
  const int o = i % O;
  const int r = i / O;
  const int n = r % 64;
  const int b = r / 64;
  const float sc = scl ? scl[o] : 1.f;
  const float sh = sft ? sft[o] : 0.f;
  float mem = 0.f;
  #pragma unroll
  for (int t = 0; t < 4; t++) {
    const size_t row = (size_t)(t * 128 + b) * 64 + n;
    float y = Y[row * O + o] * sc + sh;
    mem += (y - mem) * 0.5f;                 // TAU = 2
    float s = (mem >= th) ? 1.f : 0.f;
    out[row * O + o] = (TO)s;
    mem *= (1.f - s);
  }
}

// ---------------- spike attention, one block per (t,b,head) -----------------
// attn = q k^T * 0.125 ; y = attn v.  q,k,v binary f16 -> exact in f16 WMMA.
__global__ __launch_bounds__(128)
void k_attn(const _Float16* __restrict__ Q, const _Float16* __restrict__ K,
            const _Float16* __restrict__ V, float* __restrict__ Yo) {
  __shared__ alignas(16) _Float16 qs[64 * 32];
  __shared__ alignas(16) _Float16 ks[64 * 32];
  __shared__ alignas(16) _Float16 vt[32 * 64];   // v transposed: [d][token]
  __shared__ alignas(16) _Float16 as[64 * 64];   // attn in f16

  const int h  = blockIdx.x % 12;
  const int tb = blockIdx.x / 12;
  const size_t base = (size_t)tb * 64 * 384 + h * 32;

  for (int c = threadIdx.x; c < 256; c += 128) {
    const int row = c >> 2;
    const int col = (c & 3) * 8;
    const size_t off = base + (size_t)row * 384 + col;
    *(u32x4*)&qs[row * 32 + col] = *(const u32x4*)(Q + off);
    *(u32x4*)&ks[row * 32 + col] = *(const u32x4*)(K + off);
    const _Float16* vp = V + off;
    #pragma unroll
    for (int e = 0; e < 8; e++) vt[(col + e) * 64 + row] = vp[e];
  }
  __syncthreads();

  const int wave = threadIdx.x >> 5;
  const int lane = threadIdx.x & 31;
  const int lm   = lane & 15;
  const int half = lane >> 4;

  FragHF qa;
  { const _Float16* p = &qs[(wave * 16 + lm) * 32 + half * 8];
    qa.q[0] = *(const u32x4*)p; qa.q[1] = *(const u32x4*)(p + 16); }

  #pragma unroll
  for (int j = 0; j < 4; j++) {                 // attn: M=64(4 waves), N=64, K=32
    FragHF kb;
    const _Float16* p = &ks[(j * 16 + lm) * 32 + half * 8];
    kb.q[0] = *(const u32x4*)p; kb.q[1] = *(const u32x4*)(p + 16);
    f32x8 acc = {};
    acc = __builtin_amdgcn_wmma_f32_16x16x32_f16(false, qa.v, false, kb.v, (short)0, acc, false, false);
    #pragma unroll
    for (int r = 0; r < 8; r++) {
      const int m = wave * 16 + half * 8 + r;
      as[m * 64 + j * 16 + lm] = (_Float16)(acc[r] * 0.125f);   // SCALE, exact
    }
  }
  __syncthreads();

  #pragma unroll
  for (int j = 0; j < 2; j++) {                 // y: M=64, N=32, K=64 (2 slabs)
    f32x8 acc = {};
    #pragma unroll
    for (int kc = 0; kc < 2; kc++) {
      FragHF aa, vb;
      const _Float16* pa = &as[(wave * 16 + lm) * 64 + kc * 32 + half * 8];
      aa.q[0] = *(const u32x4*)pa; aa.q[1] = *(const u32x4*)(pa + 16);
      const _Float16* pv = &vt[(j * 16 + lm) * 64 + kc * 32 + half * 8];
      vb.q[0] = *(const u32x4*)pv; vb.q[1] = *(const u32x4*)(pv + 16);
      acc = __builtin_amdgcn_wmma_f32_16x16x32_f16(false, aa.v, false, vb.v, (short)0, acc, false, false);
    }
    #pragma unroll
    for (int r = 0; r < 8; r++) {
      const int m = wave * 16 + half * 8 + r;
      Yo[((size_t)tb * 64 + m) * 384 + h * 32 + j * 16 + lm] = acc[r];
    }
  }
}

// ---------------- depthwise 3x3 conv on spikes + BN stats -------------------
// One thread per (channel, 32-row strip): filter taps in registers, private
// stats accumulation, 2 atomics per thread (32x fewer than per-element).
__global__ __launch_bounds__(256)
void k_conv_dw(const __bf16* __restrict__ H, const float* __restrict__ Kw,
               float* __restrict__ CB, float* __restrict__ stats) {
  const int c    = blockIdx.y * 256 + threadIdx.x;   // grid.y = 3 -> 768 channels
  const int row0 = blockIdx.x * 32;                  // grid.x = 32768/32 = 1024
  float w[9];
  #pragma unroll
  for (int t = 0; t < 9; t++) w[t] = Kw[c * 9 + t];

  float s1 = 0.f, s2 = 0.f;
  for (int r = 0; r < 32; r++) {
    const int row = row0 + r;
    const int n   = row & 63;
    const int tb  = row >> 6;
    const int yy = n >> 3, xx = n & 7;
    float acc = 0.f;
    #pragma unroll
    for (int ky = 0; ky < 3; ky++) {
      const int iy = yy + ky - 1;
      if ((unsigned)iy >= 8u) continue;
      #pragma unroll
      for (int kx = 0; kx < 3; kx++) {
        const int ix = xx + kx - 1;
        if ((unsigned)ix >= 8u) continue;
        acc += (float)H[((size_t)tb * 64 + iy * 8 + ix) * 1536 + c] * w[ky * 3 + kx];
      }
    }
    CB[(size_t)row * 768 + c] = acc;
    s1 += acc; s2 += acc * acc;
  }
  atomicAdd(&stats[c],       s1);
  atomicAdd(&stats[768 + c], s2);
}

// ---------------- conv BN + LIF + gate with x2 spikes -----------------------
__global__ void k_bn_lif_gate(const float* __restrict__ CB, const float* __restrict__ scl,
                              const float* __restrict__ sft, const __bf16* __restrict__ H,
                              __bf16* __restrict__ G, int total) {
  int i = blockIdx.x * blockDim.x + threadIdx.x;
  if (i >= total) return;
  const int c = i % 768;
  const int r = i / 768;
  const int n = r % 64;
  const int b = r / 64;
  const float sc = scl[c], sh = sft[c];
  float mem = 0.f;
  #pragma unroll
  for (int t = 0; t < 4; t++) {
    const size_t row = (size_t)(t * 128 + b) * 64 + n;
    float y = CB[row * 768 + c] * sc + sh;
    mem += (y - mem) * 0.5f;
    float s = (mem >= 1.f) ? 1.f : 0.f;
    mem *= (1.f - s);
    float x2 = (float)H[row * 1536 + 768 + c];
    G[row * 768 + c] = (__bf16)(s * x2);
  }
}

// ---------------- residual add (f32 + bf16 spikes) --------------------------
__global__ void k_residual(const float* __restrict__ x, const __bf16* __restrict__ sp,
                           float* __restrict__ XR, __bf16* __restrict__ XRB, int n) {
  int i = blockIdx.x * blockDim.x + threadIdx.x;
  if (i >= n) return;
  float v = x[i] + (float)sp[i];
  XR[i]  = v;
  XRB[i] = (__bf16)v;
}

__global__ void k_final(const float* __restrict__ XR, const __bf16* __restrict__ sp,
                        float* __restrict__ out, int n) {
  int i = blockIdx.x * blockDim.x + threadIdx.x;
  if (i >= n) return;
  out[i] = XR[i] + (float)sp[i];
}

// ---------------------------------------------------------------------------
extern "C" void kernel_launch(void* const* d_in, const int* in_sizes, int n_in,
                              void* d_out, int out_size, void* d_ws, size_t ws_size,
                              hipStream_t stream) {
  (void)in_sizes; (void)n_in; (void)out_size; (void)ws_size;

  const float* x   = (const float*)d_in[0];
  const float* qw  = (const float*)d_in[1],  *qb  = (const float*)d_in[2];
  const float* qg  = (const float*)d_in[3],  *qbe = (const float*)d_in[4];
  const float* kw  = (const float*)d_in[5],  *kb  = (const float*)d_in[6];
  const float* kg  = (const float*)d_in[7],  *kbe = (const float*)d_in[8];
  const float* vw  = (const float*)d_in[9],  *vb  = (const float*)d_in[10];
  const float* vg  = (const float*)d_in[11], *vbe = (const float*)d_in[12];
  const float* pw  = (const float*)d_in[13], *pb  = (const float*)d_in[14];
  const float* pg  = (const float*)d_in[15], *pbe = (const float*)d_in[16];
  const float* f1w = (const float*)d_in[17], *f1b = (const float*)d_in[18];
  const float* f1g = (const float*)d_in[19], *f1be= (const float*)d_in[20];
  const float* f2w = (const float*)d_in[21], *f2b = (const float*)d_in[22];
  const float* f2g = (const float*)d_in[23], *f2be= (const float*)d_in[24];
  const float* dwk = (const float*)d_in[25];
  const float* dwg = (const float*)d_in[26];
  const float* dwbe= (const float*)d_in[27];

  constexpr int C = 384, HID = 1536, Ch = 768;
  constexpr int M = 4 * 128 * 64;          // 32768 rows
  const float invM = 1.0f / (float)M;

  char* wp = (char*)d_ws;
  auto take = [&](size_t bytes) { char* r = wp; wp += (bytes + 255) & ~(size_t)255; return r; };
  __bf16*   XB  = (__bf16*)  take((size_t)M * C * 2);
  __bf16*   XRB = (__bf16*)  take((size_t)M * C * 2);
  float*    XR  = (float*)   take((size_t)M * C * 4);
  _Float16* QS  = (_Float16*)take((size_t)M * C * 2);   // also reused as bf16 spike buf
  _Float16* KS  = (_Float16*)take((size_t)M * C * 2);
  _Float16* VS  = (_Float16*)take((size_t)M * C * 2);
  __bf16*   SP  = (__bf16*)  take((size_t)M * Ch * 2);  // spikes / gated
  __bf16*   HB  = (__bf16*)  take((size_t)M * HID * 2); // fc1 spikes
  float*    Y   = (float*)   take((size_t)M * HID * 4); // GEMM out; aliased by conv out
  __bf16*   WB  = (__bf16*)  take((size_t)HID * C * 2); // converted weights
  float*    stats = (float*) take(2 * HID * 4);
  float*    scl   = (float*) take(HID * 4);
  float*    sft   = (float*) take(HID * 4);

  auto cvt = [&](const float* s, __bf16* d, int n) {
    k_f32_to_bf16<<<(n + 255) / 256, 256, 0, stream>>>(s, d, n);
  };

  cvt(x, XB, M * C);

  // ---- q/k/v projections: bf16 GEMM + fused BN stats, BN+LIF -> f16 spikes
  const float* Lw[3]  = { qw, kw, vw };
  const float* Lb[3]  = { qb, kb, vb };
  const float* Lg[3]  = { qg, kg, vg };
  const float* Lbe[3] = { qbe, kbe, vbe };
  _Float16* Lo[3] = { QS, KS, VS };
  for (int i = 0; i < 3; i++) {
    hipMemsetAsync(stats, 0, 2 * C * sizeof(float), stream);
    cvt(Lw[i], WB, C * C);
    k_gemm_bf16<<<dim3(M / 128, C / 128), 128, 0, stream>>>(XB, WB, Lb[i], Y, stats, M, C, C);
    k_finalize<<<(C + 255) / 256, 256, 0, stream>>>(stats, Lg[i], Lbe[i], scl, sft, C, invM);
    const int tot = 128 * 64 * C;
    k_bn_lif<_Float16><<<(tot + 255) / 256, 256, 0, stream>>>(Y, scl, sft, Lo[i], C, 1.0f, tot);
  }

  // ---- attention (f16 WMMA, exact on binary spikes) -> Y (f32)
  k_attn<<<4 * 128 * 12, 128, 0, stream>>>(QS, KS, VS, Y);

  // ---- LIF(0.5) on attention output -> SP (bf16 spikes)
  { const int tot = 128 * 64 * C;
    k_bn_lif<__bf16><<<(tot + 255) / 256, 256, 0, stream>>>(Y, nullptr, nullptr, SP, C, 0.5f, tot); }

  // ---- p projection + BN + LIF -> spikes; residual
  hipMemsetAsync(stats, 0, 2 * C * sizeof(float), stream);
  cvt(pw, WB, C * C);
  k_gemm_bf16<<<dim3(M / 128, C / 128), 128, 0, stream>>>(SP, WB, pb, Y, stats, M, C, C);
  k_finalize<<<(C + 255) / 256, 256, 0, stream>>>(stats, pg, pbe, scl, sft, C, invM);
  { const int tot = 128 * 64 * C;
    k_bn_lif<__bf16><<<(tot + 255) / 256, 256, 0, stream>>>(Y, scl, sft, (__bf16*)QS, C, 1.0f, tot); }
  k_residual<<<(M * C + 255) / 256, 256, 0, stream>>>(x, (const __bf16*)QS, XR, XRB, M * C);

  // ---- fc1 + BN + LIF -> HB spikes (M x 1536)
  hipMemsetAsync(stats, 0, 2 * HID * sizeof(float), stream);
  cvt(f1w, WB, HID * C);
  k_gemm_bf16<<<dim3(M / 128, HID / 128), 128, 0, stream>>>(XRB, WB, f1b, Y, stats, M, C, HID);
  k_finalize<<<(HID + 255) / 256, 256, 0, stream>>>(stats, f1g, f1be, scl, sft, HID, invM);
  { const int tot = 128 * 64 * HID;
    k_bn_lif<__bf16><<<(tot + 255) / 256, 256, 0, stream>>>(Y, scl, sft, HB, HID, 1.0f, tot); }

  // ---- depthwise conv on x1 half + BN + LIF + gate with x2 -> SP (gated)
  hipMemsetAsync(stats, 0, 2 * Ch * sizeof(float), stream);
  float* CB = Y;   // alias: Y consumed, safe to reuse
  k_conv_dw<<<dim3(M / 32, Ch / 256), 256, 0, stream>>>(HB, dwk, CB, stats);
  k_finalize<<<(Ch + 255) / 256, 256, 0, stream>>>(stats, dwg, dwbe, scl, sft, Ch, invM);
  { const int tot = 128 * 64 * Ch;
    k_bn_lif_gate<<<(tot + 255) / 256, 256, 0, stream>>>(CB, scl, sft, HB, SP, tot); }

  // ---- fc2 + BN + LIF -> spikes; final residual
  hipMemsetAsync(stats, 0, 2 * C * sizeof(float), stream);
  cvt(f2w, WB, C * Ch);
  k_gemm_bf16<<<dim3(M / 128, C / 128), 128, 0, stream>>>(SP, WB, f2b, Y, stats, M, Ch, C);
  k_finalize<<<(C + 255) / 256, 256, 0, stream>>>(stats, f2g, f2be, scl, sft, C, invM);
  { const int tot = 128 * 64 * C;
    k_bn_lif<__bf16><<<(tot + 255) / 256, 256, 0, stream>>>(Y, scl, sft, (__bf16*)QS, C, 1.0f, tot); }
  k_final<<<(M * C + 255) / 256, 256, 0, stream>>>(XR, (const __bf16*)QS, (float*)d_out, M * C);
}
